// Attr_sampler_46033459479301
// MI455X (gfx1250) — matrix operand order repaired and checked by reading, not compiled, not run on hardware
//
#include <hip/hip_runtime.h>
#include <hip/hip_bf16.h>

#define N_NODES  131072
#define N_GRAPHS 256
#define IN_DIM   256
#define HID_DIM  512
#define OUT_DIM  1024     // 2*HID_DIM
#define N_EDGES  2097152
#define RATIO    0.9f
#define LN_EPS   1e-5f
#define MT       4        // M-tiles (of 16 rows) per block -> 64 nodes/block

typedef __attribute__((ext_vector_type(16))) __bf16 v16bf;
typedef __attribute__((ext_vector_type(8)))  float  v8f;

union FragAB { int4 q[2]; v16bf v; };

__device__ __forceinline__ unsigned short f2bf(float f) {
  unsigned u = __float_as_uint(f);
  unsigned r = ((u >> 16) & 1u) + 0x7FFFu;   // round-to-nearest-even
  return (unsigned short)((u + r) >> 16);
}
__device__ __forceinline__ unsigned packbf(float lo, float hi) {
  return (unsigned)f2bf(lo) | ((unsigned)f2bf(hi) << 16);
}

// ---------------------------------------------------------------------------
// Kernel 1: repack W1 (f32, [512,1024]) into bf16 WMMA B-fragments.
// B layout used here (32x16, contiguous k, same as A): lane n (0-15) holds col n, K=0..15 packed
// 2-per-dword across regs 0..7; lanes 16-31 hold K=16..31.
// ---------------------------------------------------------------------------
__global__ void pack_w1_kernel(const float* __restrict__ W1,
                               unsigned* __restrict__ wpk) {
  int t    = blockIdx.x * blockDim.x + threadIdx.x;  // 1024 tiles * 32 lanes
  int lane = t & 31;
  int tile = t >> 5;          // tile = ct*16 + k0
  int k0   = tile & 15;
  int ct   = tile >> 4;
  int n    = ct * 16 + (lane & 15);
  int hi   = lane >> 4;
  unsigned* dst = wpk + (size_t)tile * 256 + lane * 8;
#pragma unroll
  for (int r = 0; r < 8; ++r) {
    int k = k0 * 32 + 2 * r + hi * 16;
    float lo = W1[(size_t)k * OUT_DIM + n];
    float hh = W1[(size_t)(k + 1) * OUT_DIM + n];
    dst[r] = packbf(lo, hh);
  }
}

// ---------------------------------------------------------------------------
// Kernel 2: fused head.  1 block = 64 nodes (4 M-tiles), 8 waves, each wave
// owns 128 cols (8 col-tiles).  Every B fragment is loaded once and reused
// by 4 WMMAs (4x less L2 traffic for W1).  z never materialized: +b1,
// LayerNorm stats, normalize, ReLU, dot W2, sigmoid all from accumulators.
// ---------------------------------------------------------------------------
__global__ __launch_bounds__(256, 1) void head_kernel(
    const float* __restrict__ h, const unsigned* __restrict__ wpk,
    const float* __restrict__ b1, const float* __restrict__ gamma,
    const float* __restrict__ beta, const float* __restrict__ W2,
    const float* __restrict__ b2, float* __restrict__ logits) {
  __shared__ unsigned lds_a[MT * 16][264];   // 64 rows x 256 bf16-pairs (+pad)
  __shared__ float red_s[8 * 64], red_q[8 * 64], dred[8 * 64];
  __shared__ float mu_s[64], rs_s[64];

  const int tid   = threadIdx.x;
  const int lane  = tid & 31;
  const int wave  = tid >> 5;
  const int n     = lane & 15;
  const int hi    = lane >> 4;
  const int node0 = blockIdx.x * (MT * 16);

  // stage h tile [64 x 512] f32 -> packed bf16 pairs in LDS
  for (int i = tid; i < MT * 16 * 256; i += 256) {
    int row = i >> 8, pair = i & 255;
    float2 v = *(const float2*)(h + (size_t)(node0 + row) * HID_DIM + pair * 2);
    lds_a[row][pair] = packbf(v.x, v.y);
  }
  __syncthreads();

  v8f acc[MT][8];
#pragma unroll
  for (int mt = 0; mt < MT; ++mt)
#pragma unroll
    for (int ct = 0; ct < 8; ++ct)
      acc[mt][ct] = (v8f){0.f,0.f,0.f,0.f,0.f,0.f,0.f,0.f};

  const int row = n;  // A layout: lanes 0-15 -> M=lane, lanes 16-31 -> M=lane-16

  for (int k0 = 0; k0 < 16; ++k0) {
    // A fragments for the 4 M-tiles (regs 0-7: K pairs {0..7}+8hi)
    FragAB a[MT];
#pragma unroll
    for (int mt = 0; mt < MT; ++mt) {
      a[mt].q[0] = *(const int4*)&lds_a[mt * 16 + row][k0 * 16 + 8 * hi];
      a[mt].q[1] = *(const int4*)&lds_a[mt * 16 + row][k0 * 16 + 8 * hi + 4];
    }
#pragma unroll
    for (int ct = 0; ct < 8; ++ct) {
      const unsigned* wb =
          wpk + (size_t)(((wave * 8 + ct) * 16 + k0)) * 256 + lane * 8;
      FragAB b;
      b.q[0] = *(const int4*)wb;
      b.q[1] = *(const int4*)(wb + 4);
#pragma unroll
      for (int mt = 0; mt < MT; ++mt)
        acc[mt][ct] = __builtin_amdgcn_wmma_f32_16x16x32_bf16(
            false, a[mt].v, false, b.v, (short)0, acc[mt][ct], false, false);
    }
  }

  // +b1, per-lane partial sum / sumsq per row (row = mt*16 + r + 8*hi)
  const int colbase = wave * 128 + n;
  float s[MT][8], q[MT][8];
#pragma unroll
  for (int mt = 0; mt < MT; ++mt)
#pragma unroll
    for (int r = 0; r < 8; ++r) { s[mt][r] = 0.f; q[mt][r] = 0.f; }
#pragma unroll
  for (int ct = 0; ct < 8; ++ct) {
    float b1v = b1[colbase + ct * 16];
#pragma unroll
    for (int mt = 0; mt < MT; ++mt)
#pragma unroll
      for (int r = 0; r < 8; ++r) {
        float z = acc[mt][ct][r] + b1v;
        acc[mt][ct][r] = z;
        s[mt][r] += z;
        q[mt][r] += z * z;
      }
  }
  // butterfly within each 16-lane half (cols), then cross-wave via LDS
#pragma unroll
  for (int m = 1; m < 16; m <<= 1) {
#pragma unroll
    for (int mt = 0; mt < MT; ++mt)
#pragma unroll
      for (int r = 0; r < 8; ++r) {
        s[mt][r] += __shfl_xor(s[mt][r], m, 32);
        q[mt][r] += __shfl_xor(q[mt][r], m, 32);
      }
  }
  if ((lane & 15) == 0) {
#pragma unroll
    for (int mt = 0; mt < MT; ++mt)
#pragma unroll
      for (int r = 0; r < 8; ++r) {
        red_s[wave * 64 + mt * 16 + r + 8 * hi] = s[mt][r];
        red_q[wave * 64 + mt * 16 + r + 8 * hi] = q[mt][r];
      }
  }
  __syncthreads();
  if (tid < 64) {
    float S = 0.f, Q = 0.f;
    for (int w = 0; w < 8; ++w) { S += red_s[w * 64 + tid]; Q += red_q[w * 64 + tid]; }
    float mu  = S * (1.f / OUT_DIM);
    float var = Q * (1.f / OUT_DIM) - mu * mu;
    mu_s[tid] = mu;
    rs_s[tid] = rsqrtf(var + LN_EPS);
  }
  __syncthreads();

  // normalize + relu + dot with W2, reduce per row
  float d[MT][8];
#pragma unroll
  for (int mt = 0; mt < MT; ++mt)
#pragma unroll
    for (int r = 0; r < 8; ++r) d[mt][r] = 0.f;
#pragma unroll
  for (int ct = 0; ct < 8; ++ct) {
    int col = colbase + ct * 16;
    float g = gamma[col], be = beta[col], w2 = W2[col];
#pragma unroll
    for (int mt = 0; mt < MT; ++mt)
#pragma unroll
      for (int r = 0; r < 8; ++r) {
        int m = mt * 16 + r + 8 * hi;
        float zn = (acc[mt][ct][r] - mu_s[m]) * rs_s[m] * g + be;
        zn = zn > 0.f ? zn : 0.f;
        d[mt][r] += zn * w2;
      }
  }
#pragma unroll
  for (int m = 1; m < 16; m <<= 1) {
#pragma unroll
    for (int mt = 0; mt < MT; ++mt)
#pragma unroll
      for (int r = 0; r < 8; ++r) d[mt][r] += __shfl_xor(d[mt][r], m, 32);
  }
  if ((lane & 15) == 0) {
#pragma unroll
    for (int mt = 0; mt < MT; ++mt)
#pragma unroll
      for (int r = 0; r < 8; ++r)
        dred[wave * 64 + mt * 16 + r + 8 * hi] = d[mt][r];
  }
  __syncthreads();
  if (tid < 64) {
    float D = 0.f;
    for (int w = 0; w < 8; ++w) D += dred[w * 64 + tid];
    float t = D + b2[0];
    logits[node0 + tid] = 1.f / (1.f + __expf(-t));
  }
}

// ---------------------------------------------------------------------------
// Kernel 3: per-graph top-k mask by rank counting (stable: ties -> lower idx).
// ---------------------------------------------------------------------------
__device__ __forceinline__ int lower_bound_i(const int* __restrict__ b, int nn, int v) {
  int lo = 0, hi = nn;
  while (lo < hi) {
    int mid = (lo + hi) >> 1;
    if (b[mid] < v) lo = mid + 1; else hi = mid;
  }
  return lo;
}

__global__ void topk_kernel(const float* __restrict__ logits,
                            const int* __restrict__ batch,
                            float* __restrict__ mask) {
  __shared__ float sbuf[2048];
  __shared__ int se[2];
  const int g = blockIdx.x, tid = threadIdx.x;
  if (tid == 0) se[0] = lower_bound_i(batch, N_NODES, g);
  if (tid == 1) se[1] = lower_bound_i(batch, N_NODES, g + 1);
  __syncthreads();
  const int start = se[0], end = se[1], size = end - start;
  const int k = (int)ceilf(RATIO * (float)size);
  if (size <= 2048) {
    for (int i = tid; i < size; i += 256) sbuf[i] = logits[start + i];
    __syncthreads();
    for (int i = tid; i < size; i += 256) {
      float my = sbuf[i];
      int rank = 0;
      for (int j = 0; j < size; ++j) {
        float lj = sbuf[j];
        rank += (lj > my) || (lj == my && j < i);
      }
      mask[start + i] = (rank < k) ? 1.f : 0.f;
    }
  } else {  // fallback (statistically unreachable for this distribution)
    for (int i = tid; i < size; i += 256) {
      float my = logits[start + i];
      int rank = 0;
      for (int j = 0; j < size; ++j) {
        float lj = logits[start + j];
        rank += (lj > my) || (lj == my && j < i);
      }
      mask[start + i] = (rank < k) ? 1.f : 0.f;
    }
  }
}

// ---------------------------------------------------------------------------
// Kernel 4: x_new = mask ? x*logit : noise (float4), also zero edge-mask.
// ---------------------------------------------------------------------------
__global__ void xnew_kernel(const float4* __restrict__ x,
                            const float4* __restrict__ noise,
                            const float* __restrict__ logits,
                            const float* __restrict__ mask,
                            float4* __restrict__ x_new,
                            float* __restrict__ emask) {
  size_t idx = (size_t)blockIdx.x * blockDim.x + threadIdx.x;  // N*64 float4s
  int node = (int)(idx >> 6);
  float lg = logits[node];
  bool keep = (mask[node] != 0.f);
  float4 xv = x[idx], nv = noise[idx], o;
  o.x = keep ? xv.x * lg : nv.x;
  o.y = keep ? xv.y * lg : nv.y;
  o.z = keep ? xv.z * lg : nv.z;
  o.w = keep ? xv.w * lg : nv.w;
  x_new[idx] = o;
  if (idx < N_NODES) emask[idx] = 0.f;
}

// ---------------------------------------------------------------------------
// Kernel 5: edge-touch scatter (constant 1.0 -> benign race, deterministic).
// ---------------------------------------------------------------------------
__global__ void edge_kernel(const int* __restrict__ edge_index,
                            float* __restrict__ emask) {
  size_t idx = (size_t)blockIdx.x * blockDim.x + threadIdx.x;  // 2*N_EDGES
  int node = edge_index[idx];
  emask[node] = 1.f;
}

extern "C" void kernel_launch(void* const* d_in, const int* in_sizes, int n_in,
                              void* d_out, int out_size, void* d_ws, size_t ws_size,
                              hipStream_t stream) {
  (void)in_sizes; (void)n_in; (void)out_size; (void)ws_size;
  const float* h     = (const float*)d_in[0];
  const float* x     = (const float*)d_in[1];
  const float* noise = (const float*)d_in[2];
  const float* W1    = (const float*)d_in[3];
  const float* b1    = (const float*)d_in[4];
  const float* gamma = (const float*)d_in[5];
  const float* beta  = (const float*)d_in[6];
  const float* W2    = (const float*)d_in[7];
  const float* b2    = (const float*)d_in[8];
  const int* batch   = (const int*)d_in[9];
  const int* edges   = (const int*)d_in[10];

  float* out    = (float*)d_out;
  float* x_new  = out;                                  // N*256
  float* logits = out + (size_t)N_NODES * IN_DIM;       // N
  float* emask  = logits + N_NODES;                     // N

  unsigned* wpk = (unsigned*)d_ws;                      // 1 MB packed W1
  float* mask   = (float*)((char*)d_ws + (size_t)1024 * 256 * 4);  // +512 KB

  pack_w1_kernel<<<128, 256, 0, stream>>>(W1, wpk);
  head_kernel<<<N_NODES / (MT * 16), 256, 0, stream>>>(h, wpk, b1, gamma, beta, W2, b2, logits);
  topk_kernel<<<N_GRAPHS, 256, 0, stream>>>(logits, batch, mask);
  xnew_kernel<<<(N_NODES * (size_t)IN_DIM / 4) / 256, 256, 0, stream>>>(
      (const float4*)x, (const float4*)noise, logits, mask, (float4*)x_new, emask);
  edge_kernel<<<(2 * N_EDGES) / 256, 256, 0, stream>>>(edges, emask);
}
